// HTA_38714835206337
// MI455X (gfx1250) — matrix operand order repaired and checked by reading, not compiled, not run on hardware
//
#include <hip/hip_runtime.h>
#include <hip/hip_bf16.h>

// ---------------------------------------------------------------------------
// Channel-transposed attention (Restormer-style) for MI455X (gfx1250).
// bf16 storage end-to-end, f32 WMMA accumulation (v_wmma_f32_16x16x32_bf16).
// 64x128 block tile, 32x32 wave tile (4 WMMA / 8 frag ds_load_b128 per step),
// double-buffered LDS. NN A-tiles staged with GLOBAL_LOAD_ASYNC_TO_LDS_B128
// (ASYNCcnt + s_wait_asynccnt); B / transposed-A register-staged with b128
// LDS stores. Zero format-conversion VALU in the GEMM K-loop.
// ---------------------------------------------------------------------------

typedef __attribute__((ext_vector_type(16))) __bf16 v16bf;
typedef __attribute__((ext_vector_type(8)))  float  v8f;

__device__ __forceinline__ unsigned bfround(unsigned u) {
    return u + 0x7FFFu + ((u >> 16) & 1u);          // round-to-nearest-even
}
__device__ __forceinline__ unsigned short f2bfu(float f) {
    return (unsigned short)(bfround(__builtin_bit_cast(unsigned, f)) >> 16);
}
__device__ __forceinline__ unsigned pack2(float a, float b) {
    unsigned ua = bfround(__builtin_bit_cast(unsigned, a));
    unsigned ub = bfround(__builtin_bit_cast(unsigned, b));
    return (ua >> 16) | (ub & 0xFFFF0000u);
}
__device__ __forceinline__ float bf2f(unsigned short h) {
    return __builtin_bit_cast(float, (unsigned)h << 16);
}

// ---------------------------------------------------------------------------
// f32 -> bf16 bulk convert (float4 in, 4 bf16 out per thread)
// ---------------------------------------------------------------------------
__global__ void cvt_f32_bf16(const float* __restrict__ in,
                             unsigned short* __restrict__ out, int n4)
{
    int i = blockIdx.x * 256 + threadIdx.x;
    if (i >= n4) return;
    const float4 v = reinterpret_cast<const float4*>(in)[i];
    uint2 u; u.x = pack2(v.x, v.y); u.y = pack2(v.z, v.w);
    reinterpret_cast<uint2*>(out)[i] = u;
}

// ---------------------------------------------------------------------------
// Tiled bf16 WMMA GEMM.  TRANSA=false: C = A(MxK,row) * B(KxN,row)
//                        TRANSA=true : C = A^T * B, A stored (KxM,row)
// OUTBF: write C as bf16 (else f32).
// Block: 256 threads (8 waves, 2x4). Tile: 64(M) x 128(N), K-step 32.
// Per-z offsets: off = (z/zDiv)*sXo + (z%zDiv)*sXi ; split-K adds kc*sCk to C.
// All dims are multiples of the tile sizes for every stage here.
// ---------------------------------------------------------------------------
template<bool TRANSA, bool OUTBF>
__global__ __launch_bounds__(256)
void gemm_bf16_wmma(const unsigned short* __restrict__ A,
                    const unsigned short* __restrict__ B,
                    void* __restrict__ C,
                    int lda, int ldb, int ldc,
                    long long sAo, long long sAi, long long sBo, long long sBi,
                    long long sCo, long long sCi, int zDiv,
                    int kChunks, int kLen, long long sCk)
{
    __shared__ unsigned short sA[2][64][40];    // [buf][m][k], 80B rows (16B aligned)
    __shared__ unsigned short sB[2][128][40];   // [buf][n][k] (B transposed)

    const int tid  = threadIdx.x;
    const int lane = tid & 31, wave = tid >> 5;
    const int lo   = lane & 15, hi  = lane >> 4;
    const int wn   = wave & 3,  wm  = wave >> 2;

    const int z2 = blockIdx.z;
    const int z  = z2 / kChunks, kc = z2 % kChunks;
    const int zo = z / zDiv,     zi = z % zDiv;
    const unsigned short* Ab = A + zo * sAo + zi * sAi;
    const unsigned short* Bb = B + zo * sBo + zi * sBi;
    const long long coff = zo * sCo + zi * sCi + (long long)kc * sCk;

    const int m0 = blockIdx.y * 64, n0 = blockIdx.x * 128;
    const int kBase = kc * kLen, kSteps = kLen / 32;

    const int aR_nn = tid >> 2, aK_nn = (tid & 3) * 8;      // NN: A[m0+r][k0+kq8*8..+7]
    const int aM_tn = tid & 63, aK_tn = (tid >> 6) * 8;     // TN: A[k0+..][m0+m]

    // ---- NN A-tile: async DMA global -> LDS (no VGPR round trip) ----
    auto stageA_async = [&](int k0, int buf) {
        unsigned ldsAddr = (unsigned)(unsigned long long)&sA[buf][aR_nn][aK_nn];
        unsigned long long gAddr = (unsigned long long)
            (Ab + (size_t)(m0 + aR_nn) * lda + (k0 + aK_nn));
        asm volatile("global_load_async_to_lds_b128 %0, %1, off"
                     :: "v"(ldsAddr), "v"(gAddr) : "memory");
    };
    // ---- register staging for transposed A and for B ----
    uint4 aRv, bRv[2];
    auto loadA_tn = [&](int k0) {
        unsigned short t[8];
        #pragma unroll
        for (int i = 0; i < 8; ++i)
            t[i] = Ab[(size_t)(k0 + aK_tn + i) * lda + (m0 + aM_tn)];
        aRv.x = (unsigned)t[0] | ((unsigned)t[1] << 16);
        aRv.y = (unsigned)t[2] | ((unsigned)t[3] << 16);
        aRv.z = (unsigned)t[4] | ((unsigned)t[5] << 16);
        aRv.w = (unsigned)t[6] | ((unsigned)t[7] << 16);
    };
    auto storeA_tn = [&](int buf) {
        *reinterpret_cast<uint4*>(&sA[buf][aM_tn][aK_tn]) = aRv;  // ds_store_b128
    };
    auto loadB = [&](int k0) {
        #pragma unroll
        for (int j = 0; j < 2; ++j) {
            int idx = tid + j * 256;
            int n = idx & 127, k8 = (idx >> 7) * 8;         // B[k0+k8+i][n0+n]
            unsigned short t[8];
            #pragma unroll
            for (int i = 0; i < 8; ++i)
                t[i] = Bb[(size_t)(k0 + k8 + i) * ldb + (n0 + n)];
            bRv[j].x = (unsigned)t[0] | ((unsigned)t[1] << 16);
            bRv[j].y = (unsigned)t[2] | ((unsigned)t[3] << 16);
            bRv[j].z = (unsigned)t[4] | ((unsigned)t[5] << 16);
            bRv[j].w = (unsigned)t[6] | ((unsigned)t[7] << 16);
        }
    };
    auto storeB = [&](int buf) {
        #pragma unroll
        for (int j = 0; j < 2; ++j) {
            int idx = tid + j * 256;
            int n = idx & 127, k8 = (idx >> 7) * 8;
            *reinterpret_cast<uint4*>(&sB[buf][n][k8]) = bRv[j];
        }
    };

    v8f acc[2][2] = {{ {}, {} }, { {}, {} }};

    // ---- prologue: stage tile 0 ----
    if (!TRANSA) stageA_async(kBase, 0); else loadA_tn(kBase);
    loadB(kBase);
    if (TRANSA) storeA_tn(0);
    storeB(0);
    if (!TRANSA) asm volatile("s_wait_asynccnt 0" ::: "memory");
    __syncthreads();

    for (int s = 0; s < kSteps; ++s) {
        const int cur = s & 1;
        const bool more = (s + 1) < kSteps;
        if (more) {
            if (!TRANSA) stageA_async(kBase + (s + 1) * 32, cur ^ 1);
            else         loadA_tn(kBase + (s + 1) * 32);
            loadB(kBase + (s + 1) * 32);
        }

        // ---- fragments per documented 16-bit wave32 layouts ----
        const __bf16* ap0 = reinterpret_cast<const __bf16*>(&sA[cur][wm*32 + lo][0]);
        const __bf16* ap1 = reinterpret_cast<const __bf16*>(&sA[cur][wm*32 + 16 + lo][0]);
        const __bf16* bp0 = reinterpret_cast<const __bf16*>(&sB[cur][wn*32 + lo][0]);
        const __bf16* bp1 = reinterpret_cast<const __bf16*>(&sB[cur][wn*32 + 16 + lo][0]);
        v16bf a0, a1, b0, b1;
        #pragma unroll
        for (int j = 0; j < 8; ++j) {
            a0[j] = ap0[hi*8 + j];  a0[8+j] = ap0[16 + hi*8 + j];
            a1[j] = ap1[hi*8 + j];  a1[8+j] = ap1[16 + hi*8 + j];
        }
        #pragma unroll
        for (int j = 0; j < 8; ++j) {
            b0[j] = bp0[hi*8 + j];  b0[8+j] = bp0[16 + hi*8 + j];
            b1[j] = bp1[hi*8 + j];  b1[8+j] = bp1[16 + hi*8 + j];
        }

        acc[0][0] = __builtin_amdgcn_wmma_f32_16x16x32_bf16(false, a0, false, b0, (short)0, acc[0][0], false, false);
        acc[0][1] = __builtin_amdgcn_wmma_f32_16x16x32_bf16(false, a0, false, b1, (short)0, acc[0][1], false, false);
        acc[1][0] = __builtin_amdgcn_wmma_f32_16x16x32_bf16(false, a1, false, b0, (short)0, acc[1][0], false, false);
        acc[1][1] = __builtin_amdgcn_wmma_f32_16x16x32_bf16(false, a1, false, b1, (short)0, acc[1][1], false, false);

        if (more) {
            if (TRANSA) storeA_tn(cur ^ 1);
            storeB(cur ^ 1);
            if (!TRANSA) asm volatile("s_wait_asynccnt 0" ::: "memory");
        }
        __syncthreads();
    }

    // ---- epilogue (runs once; conversion cost amortized over whole K) ----
    #pragma unroll
    for (int mi = 0; mi < 2; ++mi) {
        #pragma unroll
        for (int ni = 0; ni < 2; ++ni) {
            const int col = n0 + wn*32 + ni*16 + lo;
            #pragma unroll
            for (int v = 0; v < 8; ++v) {
                int row = m0 + wm*32 + mi*16 + hi*8 + v;
                if (OUTBF)
                    ((unsigned short*)C)[coff + (size_t)row * ldc + col] = f2bfu(acc[mi][ni][v]);
                else
                    ((float*)C)[coff + (size_t)row * ldc + col] = acc[mi][ni][v];
            }
        }
    }
}

// ---------------------------------------------------------------------------
// Depthwise 3x3 conv, SAME padding, NCHW, groups = channels. bf16 in/out,
// f32 math. Memory-bound.
// ---------------------------------------------------------------------------
__global__ void dwconv3x3(const unsigned short* __restrict__ in,
                          const float* __restrict__ wdw,
                          unsigned short* __restrict__ out)
{
    size_t idx = (size_t)blockIdx.x * 256 + threadIdx.x;   // < 2*576*65536
    int w  = (int)(idx & 255);
    int hh = (int)((idx >> 8) & 255);
    size_t plane = idx >> 16;                              // b*576 + c
    int c = (int)(plane % 576);
    const float* wp = wdw + (size_t)c * 9;
    const unsigned short* ip = in + (plane << 16);
    float acc = 0.f;
    #pragma unroll
    for (int dy = -1; dy <= 1; ++dy) {
        int y = hh + dy;
        if ((unsigned)y > 255u) continue;
        #pragma unroll
        for (int dx = -1; dx <= 1; ++dx) {
            int xw = w + dx;
            if ((unsigned)xw > 255u) continue;
            acc += bf2f(ip[(size_t)y * 256 + xw]) * wp[(dy + 1) * 3 + (dx + 1)];
        }
    }
    out[idx] = f2bfu(acc);
}

// ---------------------------------------------------------------------------
// Deterministic per-column sum of squares over the (c h)=12288 axis.
// grid (16 w-chunks, 8 z, 2 sel); block 256 = 16 w-cols x 16 k-lanes.
// sq layout: [sel][z=b*4+h][w]  (2 x 8 x 256 floats)
// ---------------------------------------------------------------------------
__global__ void colsumsq(const unsigned short* __restrict__ qkv, float* __restrict__ sq)
{
    __shared__ float red[256];
    int sel = blockIdx.z, z = blockIdx.y, w0 = blockIdx.x * 16;
    int wi = threadIdx.x & 15, ki = threadIdx.x >> 4;
    int b = z >> 2, h = z & 3;
    const unsigned short* p = qkv + ((size_t)(b*576 + sel*192 + h*48) << 16) + w0 + wi;
    float acc = 0.f;
    for (int kk = ki; kk < 12288; kk += 16) {
        float v = bf2f(p[(size_t)kk * 256]);
        acc += v * v;
    }
    red[threadIdx.x] = acc; __syncthreads();
    for (int s = 128; s >= 16; s >>= 1) {
        if (threadIdx.x < s) red[threadIdx.x] += red[threadIdx.x + s];
        __syncthreads();
    }
    if (threadIdx.x < 16)
        sq[(size_t)sel * 2048 + (size_t)z * 256 + w0 + threadIdx.x] = red[threadIdx.x];
}

// ---------------------------------------------------------------------------
// Fused: sum split-K partials + scale by temperature/(||q|| ||k||) + softmax
// over axis w (dim -2).  One block per (z, v); thread t = w. bf16 output.
// ---------------------------------------------------------------------------
__global__ void softmax_scale_w(const float* __restrict__ attnP,
                                unsigned short* __restrict__ attn,
                                const float* __restrict__ qsq,
                                const float* __restrict__ ksq,
                                const float* __restrict__ temp, int kChunks)
{
    __shared__ float red[256];
    int v = blockIdx.x, z = blockIdx.y, t = threadIdx.x;
    size_t off = (size_t)z * 65536 + (size_t)t * 256 + v;
    float x = 0.f;
    for (int kc = 0; kc < kChunks; ++kc) x += attnP[off + (size_t)kc * 524288];
    float nq = fmaxf(sqrtf(qsq[(size_t)z * 256 + t]), 1e-12f);
    float nk = fmaxf(sqrtf(ksq[(size_t)z * 256 + v]), 1e-12f);
    x = x * temp[z & 3] / (nq * nk);

    red[t] = x; __syncthreads();
    for (int s = 128; s > 0; s >>= 1) {
        if (t < s) red[t] = fmaxf(red[t], red[t + s]);
        __syncthreads();
    }
    float m = red[0]; __syncthreads();
    float e = __expf(x - m);
    red[t] = e; __syncthreads();
    for (int s = 128; s > 0; s >>= 1) {
        if (t < s) red[t] += red[t + s];
        __syncthreads();
    }
    attn[off] = f2bfu(e / red[0]);
}

// ---------------------------------------------------------------------------
extern "C" void kernel_launch(void* const* d_in, const int* in_sizes, int n_in,
                              void* d_out, int out_size, void* d_ws, size_t ws_size,
                              hipStream_t stream)
{
    (void)in_sizes; (void)n_in; (void)out_size;
    const float* x      = (const float*)d_in[0];   // (2,192,256,256)
    const float* w_qkv  = (const float*)d_in[1];   // (576,192)
    const float* w_dw   = (const float*)d_in[2];   // (576,1,3,3)
    const float* w_proj = (const float*)d_in[3];   // (192,192)
    const float* temp   = (const float*)d_in[4];   // (4,1,1)
    // d_in[5] = num_heads (scalar) -> hardcoded 4

    const size_t P  = 65536;                       // h*w
    const int    KC = 8;                           // split-K chunks for scores

    // ---- workspace carve-out (256B aligned slabs) ----
    auto align256 = [](size_t v) { return (v + 255) & ~(size_t)255; };
    size_t off = 0;
    auto take = [&](size_t bytes) { size_t o = off; off += align256(bytes); return o; };
    const size_t oXbf   = take(2 * 192 * P * 2);          // x in bf16 (50 MB); reused as outv
    const size_t oWqkv  = take(576 * 192 * 2);
    const size_t oWproj = take(192 * 192 * 2);
    const size_t oQkvA  = take(2 * 576 * P * 2);          // 1x1-conv out, bf16 (151 MB)
    const size_t oQkvB  = take(2 * 576 * P * 2);          // depthwise out, bf16 (151 MB)
    const size_t oAttnP = take((size_t)KC * 8 * 256 * 256 * 4);  // f32 partials (16 MB)
    const size_t oAttn  = take((size_t)8 * 256 * 256 * 2);       // bf16 attn (1 MB)
    const size_t oSq    = take(2 * 8 * 256 * 4);
    if (ws_size < off) return;

    char* wsb = (char*)d_ws;
    unsigned short* xbf   = (unsigned short*)(wsb + oXbf);
    unsigned short* wqkvb = (unsigned short*)(wsb + oWqkv);
    unsigned short* wprjb = (unsigned short*)(wsb + oWproj);
    unsigned short* qkvA  = (unsigned short*)(wsb + oQkvA);
    unsigned short* qkvB  = (unsigned short*)(wsb + oQkvB);
    float*          attnP = (float*)(wsb + oAttnP);
    unsigned short* attn  = (unsigned short*)(wsb + oAttn);
    float*          sq    = (float*)(wsb + oSq);
    unsigned short* outv  = xbf;                   // reuse (same element count region)

    dim3 blk(256);
    const long long cP = (long long)192 * P, qP = (long long)576 * P,
                    hP = (long long)48 * P;

    // 0) one-time f32 -> bf16 conversions (producers of GEMM operands)
    cvt_f32_bf16<<<dim3((unsigned)(2 * 192 * P / 4 / 256)), blk, 0, stream>>>(x, xbf, (int)(2 * 192 * P / 4));
    cvt_f32_bf16<<<dim3(108), blk, 0, stream>>>(w_qkv, wqkvb, 576 * 192 / 4);
    cvt_f32_bf16<<<dim3(36),  blk, 0, stream>>>(w_proj, wprjb, 192 * 192 / 4);

    // 1) qkv = w_qkv @ x         M=576 N=65536 K=192  (bf16 out)
    gemm_bf16_wmma<false, true><<<dim3(512, 9, 2), blk, 0, stream>>>(
        wqkvb, xbf, qkvA, 192, (int)P, (int)P,
        0, 0, cP, 0, qP, 0, 1, 1, 192, 0);

    // 2) depthwise 3x3 (bf16 -> bf16)
    dwconv3x3<<<dim3((unsigned)(2 * 576 * P / 256)), blk, 0, stream>>>(qkvA, w_dw, qkvB);

    // 3) column sum-of-squares for q and k (deterministic, f32)
    colsumsq<<<dim3(16, 8, 2), blk, 0, stream>>>(qkvB, sq);

    // 4) score partials = Q^T K  M=N=256, K split 8 x 1536  -> attnP (f32)
    gemm_bf16_wmma<true, false><<<dim3(2, 4, 8 * KC), blk, 0, stream>>>(
        qkvB, qkvB + (size_t)192 * P, attnP, 256, 256, 256,
        qP, hP, qP, hP, (long long)4 * 65536, 65536, 4,
        KC, 12288 / KC, (long long)8 * 65536);

    // 5) reduce partials + normalize + temperature + softmax over w (bf16 out)
    softmax_scale_w<<<dim3(256, 8), blk, 0, stream>>>(attnP, attn, sq, sq + 2048, temp, KC);

    // 6) out = V @ attn          M=12288 N=256 K=256  (bf16 out)
    gemm_bf16_wmma<false, true><<<dim3(2, 192, 8), blk, 0, stream>>>(
        qkvB + (size_t)384 * P, attn, outv, 256, 256, 256,
        qP, hP, (long long)4 * 65536, 65536, cP, hP, 4, 1, 256, 0);

    // 7) final projection        M=192 N=65536 K=192  (f32 out -> d_out)
    gemm_bf16_wmma<false, false><<<dim3(512, 3, 2), blk, 0, stream>>>(
        wprjb, outv, (float*)d_out, 192, (int)P, (int)P,
        0, 0, cP, 0, cP, 0, 1, 1, 192, 0);
}
